// DynamicConv_74844100100667
// MI455X (gfx1250) — matrix operand order repaired and checked by reading, not compiled, not run on hardware
//
#include <hip/hip_runtime.h>
#include <hip/hip_bf16.h>
#include <math.h>

typedef __attribute__((ext_vector_type(16))) __bf16 v16bf;
typedef __attribute__((ext_vector_type(8)))  __bf16 v8bf;
typedef __attribute__((ext_vector_type(8)))  float  v8f;
typedef __attribute__((ext_vector_type(4)))  int    v4i;

#define BS   32
#define CIN  128
#define COUT 128
#define HWD  64
#define KEXP 8

__device__ __forceinline__ float gelu_exact(float x) {
    return 0.5f * x * (1.0f + erff(x * 0.70710678118654752f));
}

// ---------------- Kernel 1: attention MLP + softmax + agg_b ----------------
__global__ void attn_kernel(const float* __restrict__ emb,
                            const float* __restrict__ w0, const float* __restrict__ b0,
                            const float* __restrict__ w1, const float* __restrict__ b1,
                            const float* __restrict__ w2, const float* __restrict__ b2,
                            const float* __restrict__ bias,
                            float* __restrict__ att, float* __restrict__ aggb) {
    __shared__ float e[64];
    __shared__ float h[128];
    __shared__ float att_s[KEXP];
    const int b = blockIdx.x, t = threadIdx.x;

    if (t < 64) e[t] = emb[b * 64 + t];
    __syncthreads();

    float acc = b0[t];
    #pragma unroll 4
    for (int i = 0; i < 64; i++) acc += e[i] * w0[i * 128 + t];
    h[t] = gelu_exact(acc);
    __syncthreads();

    acc = b1[t];
    #pragma unroll 4
    for (int i = 0; i < 128; i++) acc += h[i] * w1[i * 128 + t];
    float v = gelu_exact(acc);
    __syncthreads();
    h[t] = v;
    __syncthreads();

    if (t < KEXP) {
        float l = b2[t];
        for (int i = 0; i < 128; i++) l += h[i] * w2[i * KEXP + t];
        att_s[t] = l;
    }
    __syncthreads();
    if (t == 0) {
        float m = att_s[0];
        for (int k = 1; k < KEXP; k++) m = fmaxf(m, att_s[k]);
        float s = 0.f;
        for (int k = 0; k < KEXP; k++) { att_s[k] = __expf(att_s[k] - m); s += att_s[k]; }
        float inv = 1.0f / s;
        for (int k = 0; k < KEXP; k++) att_s[k] *= inv;
    }
    __syncthreads();

    if (t < KEXP) att[b * KEXP + t] = att_s[t];
    float ab = 0.f;
    #pragma unroll
    for (int k = 0; k < KEXP; k++) ab += att_s[k] * bias[k * COUT + t];
    aggb[b * COUT + t] = ab;
}

// ---------------- Kernel 2: aggregate expert weights -> bf16 ----------------
// Output layout (elements): [b][cc(4)][kh(3)][kw(3)][co(128)][cim(32)]
// so the A-tile for (b, cc, kh) is one contiguous 3*128*32 bf16 chunk (24 KB).
__global__ void aggw_kernel(const float* __restrict__ att,
                            const float* __restrict__ weight,
                            __bf16* __restrict__ aggw) {
    const int idx = blockIdx.x * 256 + threadIdx.x;   // < 32*4*9*128*32 = 4718592
    const int cim  = idx & 31;
    const int co   = (idx >> 5) & 127;
    const int rem  = idx >> 12;           // (b*4 + cc)*9 + kpos
    const int kpos = rem % 9;
    const int t2   = rem / 9;
    const int cc   = t2 & 3;
    const int b    = t2 >> 2;
    const int ci   = cc * 32 + cim;

    const float* wp = weight + (size_t)co * CIN * 9 + (size_t)ci * 9 + kpos;
    float s = 0.f;
    #pragma unroll
    for (int k = 0; k < KEXP; k++)
        s += att[b * KEXP + k] * wp[(size_t)k * COUT * CIN * 9];
    aggw[idx] = (__bf16)s;
}

// ---------------- CDNA5 async staging helpers ----------------
#if __has_builtin(__builtin_amdgcn_global_load_async_to_lds_b128)
#define HAS_ASYNC_LDS 1
#else
#define HAS_ASYNC_LDS 0
#endif

// Copy one contiguous 24 KB A-tile (1536 x 16B) global -> LDS, 256 threads.
__device__ __forceinline__ void stage_A(const __bf16* __restrict__ gsrc,
                                        __bf16* dst, int t) {
#if HAS_ASYNC_LDS
    #pragma unroll
    for (int i = 0; i < 6; i++) {
        const char* g = (const char*)gsrc + (size_t)(t + i * 256) * 16;
        char*       l = (char*)dst + (size_t)(t + i * 256) * 16;
        __builtin_amdgcn_global_load_async_to_lds_b128(
            (__attribute__((address_space(1))) v4i*)g,
            (__attribute__((address_space(3))) v4i*)l,
            0, 0);
    }
#else
    const uint4* g4 = (const uint4*)gsrc;
    uint4* s4 = (uint4*)dst;
    #pragma unroll
    for (int i = 0; i < 6; i++) s4[t + i * 256] = g4[t + i * 256];
#endif
}

__device__ __forceinline__ void async_drain() {
#if HAS_ASYNC_LDS && __has_builtin(__builtin_amdgcn_s_wait_asynccnt)
    __builtin_amdgcn_s_wait_asynccnt(0);
#endif
}

// ---------------- Kernel 3: dynamic conv as 9 shifted WMMA GEMMs ----------------
// Block = (oh, b), 256 threads = 8 waves. Wave w computes co rows [16w,16w+16) x 64 ow.
// 12-stage software pipeline (stage s = ci_chunk*3 + kh), double-buffered LDS:
// async-A + VALU-converted X staged into buffer s^1 while WMMAs consume buffer s.
__global__ __launch_bounds__(256)
void conv_kernel(const float* __restrict__ x,
                 const __bf16* __restrict__ aggw,
                 const float* __restrict__ aggb,
                 float* __restrict__ out) {
    __shared__ __align__(16) __bf16 A_lds[2][3 * 128 * 32];  // [kw][co][ci32]
    __shared__ __align__(16) __bf16 X_lds[2][66 * 32];       // [colArr][ci32]

    const int oh   = blockIdx.x;
    const int b    = blockIdx.y;
    const int t    = threadIdx.x;
    const int lane = t & 31;
    const int wave = t >> 5;
    const int laneM = lane & 15;
    const int hi    = lane >> 4;      // lane group 0 / 1
    const int m0    = wave * 16;

    v8f acc[4] = {};

    // stage one halo'd input row (f32 -> bf16, transposed [col][ci]) for stage s
    auto stageX = [&](int s, int buf) {
        const int ci0 = (s / 3) * 32;
        const int row = oh - 1 + (s % 3);
        for (int idx = t; idx < 66 * 32; idx += 256) {
            const int ci  = idx / 66;
            const int col = idx - ci * 66;        // input col = col - 1
            float v = 0.f;
            if ((unsigned)row < 64u && col >= 1 && col <= 64)
                v = x[(((size_t)b * CIN + ci0 + ci) * HWD + row) * HWD + (col - 1)];
            X_lds[buf][col * 32 + ci] = (__bf16)v;
        }
    };

    // prologue: stage 0 into buffer 0
    stage_A(aggw + (size_t)(b * 12) * 12288, A_lds[0], t);
    stageX(0, 0);

    for (int s = 0; s < 12; ++s) {
        const int cur = s & 1;
        async_drain();          // own-wave async A writes to LDS complete
        __syncthreads();        // (waits DScnt) X stores + prior reads complete

        if (s + 1 < 12) {       // prefetch next stage into the other buffer
            stage_A(aggw + (size_t)(b * 12 + s + 1) * 12288, A_lds[cur ^ 1], t);
            stageX(s + 1, cur ^ 1);
        }

        #pragma unroll
        for (int kw = 0; kw < 3; kw++) {
            // A fragment, 16x32 bf16: lane group g holds K = {8g..8g+7, 16+8g..23+8g}
            const __bf16* arow = &A_lds[cur][((kw * 128) + m0 + laneM) * 32 + hi * 8];
            v8bf alo = *(const v8bf*)(arow);
            v8bf ahi = *(const v8bf*)(arow + 16);
            v16bf a = __builtin_shufflevector(alo, ahi,
                                              0, 1, 2, 3, 4, 5, 6, 7,
                                              8, 9, 10, 11, 12, 13, 14, 15);
            #pragma unroll
            for (int nt = 0; nt < 4; nt++) {
                // B fragment, 32x16 bf16: lane group g: N = laneM, K = {8g..8g+7, 16+8g..23+8g}
                const __bf16* brow = &X_lds[cur][(nt * 16 + laneM + kw) * 32 + hi * 8];
                v8bf blo = *(const v8bf*)(brow);
                v8bf bhi = *(const v8bf*)(brow + 16);
                v16bf bv = __builtin_shufflevector(blo, bhi,
                                                   0, 1, 2, 3, 4, 5, 6, 7,
                                                   8, 9, 10, 11, 12, 13, 14, 15);
                acc[nt] = __builtin_amdgcn_wmma_f32_16x16x32_bf16(
                    false, a, false, bv, (short)0, acc[nt], false, false);
            }
        }
    }

    // --- epilogue: C/D layout -> global, fused bias
    #pragma unroll
    for (int nt = 0; nt < 4; nt++) {
        const int ow = nt * 16 + laneM;
        #pragma unroll
        for (int r = 0; r < 8; r++) {
            const int co = m0 + hi * 8 + r;
            out[(((size_t)b * COUT + co) * HWD + oh) * HWD + ow] =
                acc[nt][r] + aggb[b * COUT + co];
        }
    }
}

// ---------------- launch ----------------
extern "C" void kernel_launch(void* const* d_in, const int* in_sizes, int n_in,
                              void* d_out, int out_size, void* d_ws, size_t ws_size,
                              hipStream_t stream) {
    const float* x    = (const float*)d_in[0];
    const float* emb  = (const float*)d_in[1];
    const float* w0   = (const float*)d_in[2];
    const float* b0   = (const float*)d_in[3];
    const float* w1   = (const float*)d_in[4];
    const float* b1   = (const float*)d_in[5];
    const float* w2   = (const float*)d_in[6];
    const float* b2   = (const float*)d_in[7];
    const float* wght = (const float*)d_in[8];
    const float* bias = (const float*)d_in[9];
    float* out = (float*)d_out;

    char* ws = (char*)d_ws;
    float*  att  = (float*)ws;                       //  1 KB
    float*  aggb = (float*)(ws + 1024);              // 16 KB
    __bf16* aggw = (__bf16*)(ws + 1024 + 16384);     // 9.4 MB

    attn_kernel<<<BS, 128, 0, stream>>>(emb, w0, b0, w1, b1, w2, b2, bias, att, aggb);

    const int aggw_elems = BS * 9 * COUT * CIN;      // 4718592
    aggw_kernel<<<aggw_elems / 256, 256, 0, stream>>>(att, wght, aggw);

    dim3 grid(HWD, BS);
    conv_kernel<<<grid, 256, 0, stream>>>(x, aggw, aggb, out);
}
